// OrthoLinear_73031623901961
// MI455X (gfx1250) — matrix operand order repaired and checked by reading, not compiled, not run on hardware
//
#include <hip/hip_runtime.h>
#include <hip/hip_bf16.h>

typedef _Float16 h16;
typedef __attribute__((ext_vector_type(4)))  _Float16 v4h;
typedef __attribute__((ext_vector_type(8)))  _Float16 v8h;
typedef __attribute__((ext_vector_type(16))) _Float16 v16h;
typedef __attribute__((ext_vector_type(8)))  float    v8f;

#define O_DIM  4096
#define I_DIM  4096
#define N_ROWS 8192
#define NNZ_E  838860

// ---------------------------------------------------------------------------
// Kernel 1: dequantize int4 base weights and store TRANSPOSED as f16:
//   Wt[k][o] = (nibble(base_packed[o], k) - 8) * scales[o]
// LDS-tiled 64x64 transpose so both global read and global write are coalesced.
// ---------------------------------------------------------------------------
__global__ __launch_bounds__(256)
void dequant_tr_kernel(const int* __restrict__ packed,
                       const float* __restrict__ scales,
                       h16* __restrict__ wt) {
    __shared__ __align__(16) h16 T[64][72];   // row stride 144B (16B aligned)
    const int t  = threadIdx.x;
    const int i0 = blockIdx.x * 64;           // input-col (k) tile base
    const int o0 = blockIdx.y * 64;           // output-row (o) tile base
    const int j0 = i0 >> 1;                   // packed-word base

    {   // read 8 packed words (16 weights) of one W row, scatter into LDS^T
        const int o_rel = t >> 2;
        const int jg    = t & 3;
        const int* src = packed + (size_t)(o0 + o_rel) * (I_DIM / 2) + j0 + jg * 8;
        int4 w0 = *(const int4*)(src);
        int4 w1 = *(const int4*)(src + 4);
        float s = scales[o0 + o_rel];
        int wsv[8] = {w0.x, w0.y, w0.z, w0.w, w1.x, w1.y, w1.z, w1.w};
#pragma unroll
        for (int q = 0; q < 8; ++q) {
            float lo = (float)((wsv[q] & 0xF) - 8) * s;        // even input col
            float hi = (float)(((wsv[q] >> 4) & 0xF) - 8) * s; // odd input col
            int ir = jg * 16 + 2 * q;
            T[ir][o_rel]     = (h16)lo;
            T[ir + 1][o_rel] = (h16)hi;
        }
    }
    __syncthreads();
    {   // coalesced b128 write-out of the transposed tile
        const int i_rel = t >> 2;
        const int part  = t & 3;
        v8h a = *(const v8h*)&T[i_rel][part * 16];
        v8h b = *(const v8h*)&T[i_rel][part * 16 + 8];
        h16* dst = wt + (size_t)(i0 + i_rel) * O_DIM + o0 + part * 16;
        *(v8h*)dst       = a;
        *(v8h*)(dst + 8) = b;
    }
}

// ---------------------------------------------------------------------------
// Kernel 2: COO scatter-add of fp16 residuals into Wt via packed f16 atomics
// ---------------------------------------------------------------------------
__global__ __launch_bounds__(256)
void scatter_kernel(const h16* __restrict__ vals,
                    const int* __restrict__ rows,
                    const int* __restrict__ cols,
                    h16* __restrict__ wt) {
    int e = blockIdx.x * blockDim.x + threadIdx.x;
    if (e >= NNZ_E) return;
    int r = rows[e];
    int c = cols[e];
    union { h16 h; unsigned short u; } cv;
    cv.h = vals[e];
    unsigned idx  = (unsigned)c * (unsigned)O_DIM + (unsigned)r;  // Wt[c][r]
    unsigned pair = idx & ~1u;                                    // 4B aligned
    unsigned data = (idx & 1u) ? ((unsigned)cv.u << 16) : (unsigned)cv.u;
    unsigned long long addr =
        (unsigned long long)(uintptr_t)wt + (unsigned long long)pair * 2ull;
    asm volatile("global_atomic_pk_add_f16 %0, %1, off"
                 :: "v"(addr), "v"(data) : "memory");
}

// ---------------------------------------------------------------------------
// Kernel 3: x f32 -> f16 (xh 64MiB + Wt 32MiB both resident in 192MB L2)
// ---------------------------------------------------------------------------
__global__ __launch_bounds__(256)
void cvt_x_kernel(const float* __restrict__ x, h16* __restrict__ xh) {
    int i = blockIdx.x * blockDim.x + threadIdx.x;   // one float4 per thread
    float4 v = ((const float4*)x)[i];
    v4h o;
    o[0] = (h16)v.x; o[1] = (h16)v.y; o[2] = (h16)v.z; o[3] = (h16)v.w;
    ((v4h*)xh)[i] = o;
}

// ---------------------------------------------------------------------------
// Kernel 4: tiled WMMA GEMM  out[n][o] = sum_k xh[n][k] * Wt[k][o]
// BM=128, BN=128, BK=64; 8 wave32s; 2x4 wmma accumulators per wave.
// Double-buffered LDS filled with CDNA5 async memory->LDS copies (ASYNCcnt).
// ---------------------------------------------------------------------------
#define BM 128
#define BN 128
#define BK 64
#define LDA 72    // 144B row stride
#define LDB 136   // 272B row stride
#define KITERS (I_DIM / BK)

// 4 async b128 copies: INST_OFFSET applies to BOTH lds and global address
// (CDNA5 ISA), so one 64B chunk per thread per call.
__device__ __forceinline__ void async_fill64(unsigned lds, unsigned long long ga) {
    asm volatile(
        "global_load_async_to_lds_b128 %0, %1, off\n\t"
        "global_load_async_to_lds_b128 %0, %1, off offset:16\n\t"
        "global_load_async_to_lds_b128 %0, %1, off offset:32\n\t"
        "global_load_async_to_lds_b128 %0, %1, off offset:48"
        :: "v"(lds), "v"(ga) : "memory");
}

__global__ __launch_bounds__(256)
void gemm_wmma_kernel(const h16* __restrict__ xh,
                      const h16* __restrict__ wt,
                      float* __restrict__ out) {
    __shared__ __align__(16) h16 As[2][BM * LDA];  // [buf][m][k]
    __shared__ __align__(16) h16 Bs[2][BK * LDB];  // [buf][k][o]

    const int t    = threadIdx.x;
    const int lane = t & 31;
    const int wv   = t >> 5;
    const int wm   = wv & 3;    // wave m-group: rows [wm*32, wm*32+32)
    const int wn   = wv >> 2;   // wave n-group: cols [wn*64, wn*64+64)
    const int gm0  = blockIdx.y * BM;
    const int gn0  = blockIdx.x * BN;

    v8f acc[2][4];
#pragma unroll
    for (int i = 0; i < 2; ++i)
#pragma unroll
        for (int j = 0; j < 4; ++j)
#pragma unroll
            for (int q = 0; q < 8; ++q) acc[i][j][q] = 0.0f;

    // per-thread fill assignment: one 64B chunk of A and one of B per K-tile
    const int am  = t >> 1, ap = (t & 1) * 32;   // A row am, halves [ap,ap+32)
    const int bkr = t >> 2, bp = (t & 3) * 32;   // B k-row bkr, halves [bp,bp+32)
    unsigned long long aga =
        (unsigned long long)(uintptr_t)(xh + (size_t)(gm0 + am) * I_DIM + ap);
    unsigned long long bga =
        (unsigned long long)(uintptr_t)(wt + (size_t)bkr * O_DIM + gn0 + bp);
    unsigned alds[2], blds[2];
    alds[0] = (unsigned)(uintptr_t)&As[0][am * LDA + ap];
    alds[1] = (unsigned)(uintptr_t)&As[1][am * LDA + ap];
    blds[0] = (unsigned)(uintptr_t)&Bs[0][bkr * LDB + bp];
    blds[1] = (unsigned)(uintptr_t)&Bs[1][bkr * LDB + bp];

    const int lrow = lane & 15;
    const int sel  = (lane >> 4) * 8;   // A frag K-phase per ISA layout

    // prologue: fire async fill of buffer 0
    async_fill64(alds[0], aga);
    async_fill64(blds[0], bga);

    for (int it = 0; it < KITERS; ++it) {
        const int buf = it & 1;
        // our async copies for `buf` complete, then sync all waves
        asm volatile("s_wait_asynccnt 0x0" ::: "memory");
        __syncthreads();

        if (it + 1 < KITERS) {
            aga += (unsigned long long)BK * 2;                 // +128 B
            bga += (unsigned long long)BK * O_DIM * 2;         // +512 KiB
            async_fill64(alds[buf ^ 1], aga);
            async_fill64(blds[buf ^ 1], bga);
            __builtin_prefetch((const void*)(uintptr_t)(bga + (unsigned long long)BK * O_DIM * 2), 0, 1);
        }

        const h16* Ab = As[buf];
        const h16* Bb = Bs[buf];
#pragma unroll
        for (int ks = 0; ks < BK; ks += 32) {
            // A fragments: 16x32 f16. lane<16: M=lane, K={0..7,16..23};
            // lane>=16: M=lane-16, K={8..15,24..31}   (ISA 7.12.2)
            v16h af[2];
#pragma unroll
            for (int mt = 0; mt < 2; ++mt) {
                const h16* p = Ab + (wm * 32 + mt * 16 + lrow) * LDA + ks + sel;
                v8h lo = *(const v8h*)p;
                v8h hi = *(const v8h*)(p + 16);
#pragma unroll
                for (int q = 0; q < 8; ++q) { af[mt][q] = lo[q]; af[mt][q + 8] = hi[q]; }
            }
            // B fragments: 32x16 f16. lane holds N = lane&15, K = {sel..sel+7, sel+16..sel+23}.
            v16h bf[4];
#pragma unroll
            for (int nt = 0; nt < 4; ++nt) {
                const h16* p = Bb + (ks + sel) * LDB + wn * 64 + nt * 16 + lrow;
#pragma unroll
                for (int q = 0; q < 8; ++q) { bf[nt][q] = p[q * LDB]; bf[nt][q + 8] = p[(q + 16) * LDB]; }
            }
#pragma unroll
            for (int mt = 0; mt < 2; ++mt)
#pragma unroll
                for (int nt = 0; nt < 4; ++nt)
                    acc[mt][nt] = __builtin_amdgcn_wmma_f32_16x16x32_f16(
                        false, af[mt], false, bf[nt],
                        (short)0, acc[mt][nt], false, false);
        }
    }

    // Epilogue: D layout (ISA 7.12.2): VGPR r -> M = r (+8 for lanes 16..31),
    // N = lane&15.
    const int ncol = lane & 15;
    const int mb   = (lane >> 4) * 8;
#pragma unroll
    for (int mt = 0; mt < 2; ++mt)
#pragma unroll
        for (int nt = 0; nt < 4; ++nt) {
            float* op = out + (size_t)(gm0 + wm * 32 + mt * 16 + mb) * O_DIM
                            + gn0 + wn * 64 + nt * 16 + ncol;
#pragma unroll
            for (int r = 0; r < 8; ++r) op[(size_t)r * O_DIM] = acc[mt][nt][r];
        }
}

// ---------------------------------------------------------------------------
extern "C" void kernel_launch(void* const* d_in, const int* in_sizes, int n_in,
                              void* d_out, int out_size, void* d_ws, size_t ws_size,
                              hipStream_t stream) {
    const float* x      = (const float*)d_in[0];
    const int*   packed = (const int*)d_in[1];
    const float* scales = (const float*)d_in[2];
    const h16*   vals   = (const h16*)d_in[3];
    const int*   rows   = (const int*)d_in[4];
    const int*   cols   = (const int*)d_in[5];
    float*       out    = (float*)d_out;

    const size_t wt_bytes = (size_t)O_DIM * I_DIM * sizeof(h16);   // 32 MiB
    const size_t xh_bytes = (size_t)N_ROWS * I_DIM * sizeof(h16);  // 64 MiB
    if (ws_size < wt_bytes + xh_bytes) return;  // workspace too small

    h16* wt = (h16*)d_ws;
    h16* xh = (h16*)((char*)d_ws + wt_bytes);

    dequant_tr_kernel<<<dim3(I_DIM / 64, O_DIM / 64), 256, 0, stream>>>(packed, scales, wt);
    scatter_kernel<<<(NNZ_E + 255) / 256, 256, 0, stream>>>(vals, rows, cols, wt);
    cvt_x_kernel<<<((size_t)N_ROWS * I_DIM / 4 + 255) / 256, 256, 0, stream>>>(x, xh);
    gemm_wmma_kernel<<<dim3(O_DIM / BN, N_ROWS / BM), 256, 0, stream>>>(xh, wt, out);
}
